// HelicalAttention_87187836108808
// MI455X (gfx1250) — compile-verified, matrix operand order checked by reading, not run on hardware
//
#include <hip/hip_runtime.h>
#include <hip/hip_bf16.h>
#include <stdint.h>

// Problem constants (match reference)
#define Bn  2
#define Sq  2048
#define Ed  1024
#define Hn  16
#define HDd 64

typedef __attribute__((ext_vector_type(16))) __bf16 bf16x16;
typedef __attribute__((ext_vector_type(8)))  float  f32x8;

union FragA { bf16x16 v; uint4 u[2]; };

__device__ __forceinline__ unsigned short f2bf(float x) {
  union { float f; unsigned int u; } v; v.f = x;
  unsigned int r = v.u + 0x7FFFu + ((v.u >> 16) & 1u);
  return (unsigned short)(r >> 16);
}
__device__ __forceinline__ float bf2f(unsigned short x) {
  union { unsigned int u; float f; } v; v.u = ((unsigned int)x) << 16;
  return v.f;
}

// Issue one 16B async copy global -> LDS (CDNA5 GLOBAL_LOAD_ASYNC_TO_LDS_B128,
// tracked by ASYNCcnt; no VGPR round-trip). ldsaddr = low 32 bits of the
// generic pointer (generic LDS addrs are {aperture, lds_offset[31:0]}).
__device__ __forceinline__ void async_copy_b128(void* lds_dst, const void* gsrc) {
  unsigned ldsaddr = (unsigned)(uintptr_t)lds_dst;
  asm volatile("global_load_async_to_lds_b128 %0, %1, off"
               :: "v"(ldsaddr), "v"(gsrc) : "memory");
}
__device__ __forceinline__ void wait_async0() {
  asm volatile("s_wait_asynccnt 0x0" ::: "memory");
}

// ---------------------------------------------------------------------------
// fp32 -> bf16 elementwise convert
// ---------------------------------------------------------------------------
__global__ __launch_bounds__(256) void k_f32_to_bf16(
    const float* __restrict__ s, unsigned short* __restrict__ d, int n) {
  int i = blockIdx.x * 256 + threadIdx.x;
  if (i < n) d[i] = f2bf(s[i]);
}

// ---------------------------------------------------------------------------
// NT GEMM: C[M,N] = A[M,K] * W[N,K]^T + bias, bf16 inputs, f32 accumulate.
// Block tile 128(M) x 64(N), 8 waves; each wave: 16 rows x 64 cols (4 C tiles).
// W tile staged to LDS via async-to-LDS, shared by all 8 waves. K step = 32.
// ---------------------------------------------------------------------------
__global__ __launch_bounds__(256) void k_gemm_nt(
    const unsigned short* __restrict__ A, const unsigned short* __restrict__ W,
    const float* __restrict__ bias, float* __restrict__ Cf,
    unsigned short* __restrict__ Cb, int M, int N, int K) {
  __shared__ unsigned short Bs[64][40];  // 64 rows(N) x 32(K), +8 pad

  const int tid  = threadIdx.x;
  const int lane = tid & 31;
  const int wave = tid >> 5;
  const int m0 = blockIdx.x * 128;
  const int n0 = blockIdx.y * 64;
  const int ln16  = lane & 15;
  const int khalf = (lane >> 4) * 8;
  const int arow  = m0 + wave * 16 + ln16;

  f32x8 acc[4] = {};

  for (int kb = 0; kb < K; kb += 32) {
    __syncthreads();
    {  // async stage W tile: 64x32 bf16 = 4KB, 256 lanes x 16B, direct to LDS
      int r = tid >> 2, c = (tid & 3) * 8;
      async_copy_b128(&Bs[r][c], &W[(size_t)(n0 + r) * K + kb + c]);
    }
    wait_async0();
    __syncthreads();

    FragA a;  // A-frag: lane = row ln16, K = khalf..+7 and 16+khalf..+7
    a.u[0] = *(const uint4*)&A[(size_t)arow * K + kb + khalf];
    a.u[1] = *(const uint4*)&A[(size_t)arow * K + kb + khalf + 16];

#pragma unroll
    for (int nt = 0; nt < 4; ++nt) {
      FragA b;  // B-frag: lane = col n, same K packing
      b.u[0] = *(const uint4*)&Bs[nt * 16 + ln16][khalf];
      b.u[1] = *(const uint4*)&Bs[nt * 16 + ln16][khalf + 16];
      acc[nt] = __builtin_amdgcn_wmma_f32_16x16x32_bf16(
          false, a.v, false, b.v, (short)0, acc[nt], false, false);
    }
  }

#pragma unroll
  for (int nt = 0; nt < 4; ++nt) {
#pragma unroll
    for (int v = 0; v < 8; ++v) {
      int row = m0 + wave * 16 + v + ((lane >> 4) << 3);
      int col = n0 + nt * 16 + ln16;
      float val = acc[nt][v] + bias[col];
      if (Cb) Cb[(size_t)row * N + col] = f2bf(val);
      else    Cf[(size_t)row * N + col] = val;
    }
  }
}

// ---------------------------------------------------------------------------
// Rearrange: Qcat[...,0:64] = qh/16, Kcat[...,0:64] = kh, Vt = vh^T per head.
// ---------------------------------------------------------------------------
__global__ __launch_bounds__(256) void k_prep_qkv(
    const unsigned short* __restrict__ Qb, const unsigned short* __restrict__ Kb,
    const unsigned short* __restrict__ Vb, unsigned short* __restrict__ Qcat,
    unsigned short* __restrict__ Kcat, unsigned short* __restrict__ Vt) {
  int i = blockIdx.x * 256 + threadIdx.x;  // over B*H*S*HD = 2^22
  int d = i & 63;
  int s = (i >> 6) & (Sq - 1);
  int h = (i >> 17) & (Hn - 1);
  int b = i >> 21;
  size_t eidx = ((size_t)(b * Sq + s)) * Ed + h * 64 + d;
  size_t hrow = (size_t)(b * Hn + h) * Sq + s;
  Qcat[hrow * 128 + d] = f2bf(bf2f(Qb[eidx]) * 0.0625f);  // 0.5/sqrt(64)
  Kcat[hrow * 128 + d] = Kb[eidx];
  Vt[((size_t)(b * Hn + h) * 64 + d) * Sq + s] = Vb[eidx];
}

// ---------------------------------------------------------------------------
// Helix projection per (b,h,s,pair): two 64-deep dots + pair L2 normalize.
// ---------------------------------------------------------------------------
__global__ __launch_bounds__(256) void k_helix(
    const unsigned short* __restrict__ Qb, const unsigned short* __restrict__ Kb,
    const float* __restrict__ Whel, const float* __restrict__ bhel,
    float* __restrict__ qn, float* __restrict__ kn) {
  int i = blockIdx.x * 256 + threadIdx.x;  // over B*H*S*P = 2^21
  int p = i & 31;
  int s = (i >> 5) & (Sq - 1);
  int h = (i >> 16) & (Hn - 1);
  int b = i >> 20;
  const unsigned short* q = Qb + ((size_t)(b * Sq + s)) * Ed + h * 64;
  const unsigned short* k = Kb + ((size_t)(b * Sq + s)) * Ed + h * 64;
  const float* w0 = Whel + ((size_t)h * 64 + 2 * p) * 64;
  const float* w1 = w0 + 64;
  float qa0 = bhel[h * 64 + 2 * p], qa1 = bhel[h * 64 + 2 * p + 1];
  float ka0 = qa0, ka1 = qa1;
#pragma unroll 8
  for (int d = 0; d < 64; ++d) {
    float qd = bf2f(q[d]), kd = bf2f(k[d]);
    qa0 += qd * w0[d]; qa1 += qd * w1[d];
    ka0 += kd * w0[d]; ka1 += kd * w1[d];
  }
  float qden = fmaxf(sqrtf(qa0 * qa0 + qa1 * qa1), 1e-12f);
  float kden = fmaxf(sqrtf(ka0 * ka0 + ka1 * ka1), 1e-12f);
  size_t o = ((size_t)(b * Hn + h) * Sq + s) * 64 + 2 * p;
  qn[o] = qa0 / qden; qn[o + 1] = qa1 / qden;
  kn[o] = ka0 / kden; kn[o + 1] = ka1 / kden;
}

// ---------------------------------------------------------------------------
// Scramble gather: (B,S,2)->transpose->reshape per (b,h,p), into Qcat/Kcat hi half.
// ---------------------------------------------------------------------------
__global__ __launch_bounds__(256) void k_scramble(
    const float* __restrict__ qn, const float* __restrict__ kn,
    unsigned short* __restrict__ Qcat, unsigned short* __restrict__ Kcat) {
  int i = blockIdx.x * 256 + threadIdx.x;  // over B*H*S*P
  int p = i & 31;
  int s = (i >> 5) & (Sq - 1);
  int h = (i >> 16) & (Hn - 1);
  int b = i >> 20;
  size_t base = (size_t)(b * Hn + h) * Sq;
#pragma unroll
  for (int c = 0; c < 2; ++c) {
    int f  = 2 * s + c;
    int cc = (f >= Sq) ? 1 : 0;
    int t  = f - cc * Sq;
    float qv = qn[(base + t) * 64 + 2 * p + cc];
    float kv = kn[(base + t) * 64 + 2 * p + cc];
    size_t o = (base + s) * 128 + 64 + 2 * p + c;
    Qcat[o] = f2bf(qv * (1.0f / 64.0f));  // 0.5/P
    Kcat[o] = f2bf(kv);
  }
}

// ---------------------------------------------------------------------------
// Flash attention: per block 128 query rows of one (b,h); 8 waves x 16 rows.
// d_qk = 128 (head 64 + helix 64, scales pre-folded), d_v = 64.
// K/V tiles (32 keys) async-staged to LDS; online softmax; P re-laid out via LDS.
// ---------------------------------------------------------------------------
__global__ __launch_bounds__(256) void k_flash_attn(
    const unsigned short* __restrict__ Qcat, const unsigned short* __restrict__ Kcat,
    const unsigned short* __restrict__ Vt, unsigned short* __restrict__ AO) {
  __shared__ unsigned short Ks[32][136];    // 32 keys x 128 feat (+8 pad)
  __shared__ unsigned short Vs[64][40];     // 64 feat x 32 keys (+8 pad)
  __shared__ unsigned short Ps[8][16][40];  // per-wave P tile 16x32 (+8 pad)

  const int tid  = threadIdx.x;
  const int lane = tid & 31;
  const int wave = tid >> 5;
  const int h = blockIdx.y, b = blockIdx.z;
  const int m0 = blockIdx.x * 128;
  const size_t bh = (size_t)(b * Hn + h);
  const unsigned short* Qp = Qcat + bh * Sq * 128;
  const unsigned short* Kp = Kcat + bh * Sq * 128;
  const unsigned short* Vp = Vt + bh * 64 * Sq;

  const int ln16  = lane & 15;
  const int khalf = (lane >> 4) * 8;

  FragA aq[4];  // this wave's 16 Q rows, 128 features = 4 K-blocks of 32
  {
    const unsigned short* qr = Qp + (size_t)(m0 + wave * 16 + ln16) * 128;
#pragma unroll
    for (int j = 0; j < 4; ++j) {
      aq[j].u[0] = *(const uint4*)&qr[j * 32 + khalf];
      aq[j].u[1] = *(const uint4*)&qr[j * 32 + khalf + 16];
    }
  }

  f32x8 acc[4] = {};
  float m_r[8], l_r[8];
#pragma unroll
  for (int v = 0; v < 8; ++v) { m_r[v] = -1e30f; l_r[v] = 0.0f; }

  for (int kb = 0; kb < Sq; kb += 32) {
    __syncthreads();
#pragma unroll
    for (int it = 0; it < 2; ++it) {  // async stage K tile: 32x128 bf16 = 8KB
      int c = tid + it * 256;
      int r = c >> 4, ch = (c & 15) * 8;
      async_copy_b128(&Ks[r][ch], &Kp[(size_t)(kb + r) * 128 + ch]);
    }
    {  // async stage V tile: 64x32 bf16 = 4KB (feature-major => contiguous keys)
      int r = tid >> 2, ch = (tid & 3) * 8;
      async_copy_b128(&Vs[r][ch], &Vp[(size_t)r * Sq + kb + ch]);
    }
    wait_async0();
    __syncthreads();

    // scores: two 16-key tiles, each K=128 -> 4 chained WMMAs
    f32x8 sc[2];
#pragma unroll
    for (int kt = 0; kt < 2; ++kt) {
      f32x8 s = {};
#pragma unroll
      for (int j = 0; j < 4; ++j) {
        FragA bf;
        bf.u[0] = *(const uint4*)&Ks[kt * 16 + ln16][j * 32 + khalf];
        bf.u[1] = *(const uint4*)&Ks[kt * 16 + ln16][j * 32 + khalf + 16];
        s = __builtin_amdgcn_wmma_f32_16x16x32_bf16(
            false, aq[j].v, false, bf.v, (short)0, s, false, false);
      }
      sc[kt] = s;
    }

    // online softmax; row r = v + 8*(lane/16) lives across a 16-lane half
#pragma unroll
    for (int v = 0; v < 8; ++v) {
      float t = fmaxf(sc[0][v], sc[1][v]);
      t = fmaxf(t, __shfl_xor(t, 1, 32));
      t = fmaxf(t, __shfl_xor(t, 2, 32));
      t = fmaxf(t, __shfl_xor(t, 4, 32));
      t = fmaxf(t, __shfl_xor(t, 8, 32));
      float mn  = fmaxf(m_r[v], t);
      float rsc = __expf(m_r[v] - mn);
      float p0  = __expf(sc[0][v] - mn);
      float p1  = __expf(sc[1][v] - mn);
      float rs  = p0 + p1;
      rs += __shfl_xor(rs, 1, 32);
      rs += __shfl_xor(rs, 2, 32);
      rs += __shfl_xor(rs, 4, 32);
      rs += __shfl_xor(rs, 8, 32);
      l_r[v] = l_r[v] * rsc + rs;
      m_r[v] = mn;
#pragma unroll
      for (int nt = 0; nt < 4; ++nt) acc[nt][v] *= rsc;
      int prow = v + ((lane >> 4) << 3);  // C-layout row
      Ps[wave][prow][ln16]      = f2bf(p0);
      Ps[wave][prow][16 + ln16] = f2bf(p1);
    }
    // re-layout P (C-frag -> A-frag) via per-wave LDS; DS ops of one wave are
    // in-order, but the transpose crosses lanes, so wait for the stores.
    asm volatile("s_wait_dscnt 0" ::: "memory");

    FragA pf;
    pf.u[0] = *(const uint4*)&Ps[wave][ln16][khalf];
    pf.u[1] = *(const uint4*)&Ps[wave][ln16][khalf + 16];
#pragma unroll
    for (int nt = 0; nt < 4; ++nt) {
      FragA vf;  // B-frag from Vt: lane = feature col, contiguous in keys(K)
      vf.u[0] = *(const uint4*)&Vs[nt * 16 + ln16][khalf];
      vf.u[1] = *(const uint4*)&Vs[nt * 16 + ln16][khalf + 16];
      acc[nt] = __builtin_amdgcn_wmma_f32_16x16x32_bf16(
          false, pf.v, false, vf.v, (short)0, acc[nt], false, false);
    }
  }

#pragma unroll
  for (int nt = 0; nt < 4; ++nt) {
#pragma unroll
    for (int v = 0; v < 8; ++v) {
      int row = m0 + wave * 16 + v + ((lane >> 4) << 3);
      int col = h * 64 + nt * 16 + ln16;
      AO[((size_t)b * Sq + row) * Ed + col] = f2bf(acc[nt][v] / l_r[v]);
    }
  }
}

// ---------------------------------------------------------------------------
// Orchestration
// ---------------------------------------------------------------------------
extern "C" void kernel_launch(void* const* d_in, const int* in_sizes, int n_in,
                              void* d_out, int out_size, void* d_ws, size_t ws_size,
                              hipStream_t stream) {
  (void)in_sizes; (void)n_in; (void)out_size; (void)ws_size;
  const float* x    = (const float*)d_in[0];
  const float* Wq   = (const float*)d_in[1];
  const float* bq   = (const float*)d_in[2];
  const float* Wk   = (const float*)d_in[3];
  const float* bk   = (const float*)d_in[4];
  const float* Wv   = (const float*)d_in[5];
  const float* bv   = (const float*)d_in[6];
  const float* Wo   = (const float*)d_in[7];
  const float* bo   = (const float*)d_in[8];
  const float* Whel = (const float*)d_in[9];
  const float* bhel = (const float*)d_in[10];

  char* ws = (char*)d_ws;
  size_t off = 0;
  auto take = [&](size_t bytes) { void* p = ws + off; off += bytes; return p; };

  const size_t SZ_XE  = (size_t)Bn * Sq * Ed;  // 4M elems
  const size_t SZ_W   = (size_t)Ed * Ed;       // 1M elems

  unsigned short* Xbf  = (unsigned short*)take(SZ_XE * 2);
  unsigned short* Wqb  = (unsigned short*)take(SZ_W * 2);
  unsigned short* Wkb  = (unsigned short*)take(SZ_W * 2);
  unsigned short* Wvb  = (unsigned short*)take(SZ_W * 2);
  unsigned short* Wob  = (unsigned short*)take(SZ_W * 2);
  unsigned short* Qbf  = (unsigned short*)take(SZ_XE * 2);
  unsigned short* Kbf  = (unsigned short*)take(SZ_XE * 2);
  unsigned short* Vbf  = (unsigned short*)take(SZ_XE * 2);
  float*          qnw  = (float*)take((size_t)Bn * Hn * Sq * 64 * 4);
  float*          knw  = (float*)take((size_t)Bn * Hn * Sq * 64 * 4);
  unsigned short* Qcat = (unsigned short*)take((size_t)Bn * Hn * Sq * 128 * 2);
  unsigned short* Kcat = (unsigned short*)take((size_t)Bn * Hn * Sq * 128 * 2);
  unsigned short* Vtb  = (unsigned short*)take((size_t)Bn * Hn * 64 * Sq * 2);
  unsigned short* AO   = (unsigned short*)take(SZ_XE * 2);

  // 1) converts
  k_f32_to_bf16<<<(int)((SZ_XE + 255) / 256), 256, 0, stream>>>(x, Xbf, (int)SZ_XE);
  k_f32_to_bf16<<<(int)((SZ_W + 255) / 256), 256, 0, stream>>>(Wq, Wqb, (int)SZ_W);
  k_f32_to_bf16<<<(int)((SZ_W + 255) / 256), 256, 0, stream>>>(Wk, Wkb, (int)SZ_W);
  k_f32_to_bf16<<<(int)((SZ_W + 255) / 256), 256, 0, stream>>>(Wv, Wvb, (int)SZ_W);
  k_f32_to_bf16<<<(int)((SZ_W + 255) / 256), 256, 0, stream>>>(Wo, Wob, (int)SZ_W);

  // 2) Q/K/V projections (bf16 out)
  dim3 gg(Bn * Sq / 128, Ed / 64);
  k_gemm_nt<<<gg, 256, 0, stream>>>(Xbf, Wqb, bq, nullptr, Qbf, Bn * Sq, Ed, Ed);
  k_gemm_nt<<<gg, 256, 0, stream>>>(Xbf, Wkb, bk, nullptr, Kbf, Bn * Sq, Ed, Ed);
  k_gemm_nt<<<gg, 256, 0, stream>>>(Xbf, Wvb, bv, nullptr, Vbf, Bn * Sq, Ed, Ed);

  // 3) feature assembly
  int n1 = Bn * Hn * Sq * HDd;  // 2^22
  k_prep_qkv<<<n1 / 256, 256, 0, stream>>>(Qbf, Kbf, Vbf, Qcat, Kcat, Vtb);
  int n2 = Bn * Hn * Sq * 32;   // 2^21
  k_helix<<<n2 / 256, 256, 0, stream>>>(Qbf, Kbf, Whel, bhel, qnw, knw);
  k_scramble<<<n2 / 256, 256, 0, stream>>>(qnw, knw, Qcat, Kcat);

  // 4) fused attention
  dim3 ga(Sq / 128, Hn, Bn);
  k_flash_attn<<<ga, 256, 0, stream>>>(Qcat, Kcat, Vtb, AO);

  // 5) output projection (fp32 out)
  k_gemm_nt<<<gg, 256, 0, stream>>>(AO, Wob, bo, (float*)d_out, nullptr, Bn * Sq, Ed, Ed);
}
